// CRFClassifier_5566277616318
// MI455X (gfx1250) — matrix operand order repaired and checked by reading, not compiled, not run on hardware
//
#include <hip/hip_runtime.h>
#include <hip/hip_bf16.h>

#define Bn 64
#define Sn 512
#define Hn 1024
#define Tn 21
#define TP 32   // padded tag dimension (two 16-wide WMMA N-tiles)

typedef __attribute__((ext_vector_type(16))) __bf16        v16bf;
typedef __attribute__((ext_vector_type(8)))  float         v8f;
typedef __attribute__((ext_vector_type(4)))  unsigned int  u32x4;
typedef __attribute__((ext_vector_type(8)))  int           i32x8;
typedef __attribute__((ext_vector_type(4)))  int           i32x4;

#if defined(__has_builtin)
#if __has_builtin(__builtin_amdgcn_tensor_load_to_lds) && __has_builtin(__builtin_amdgcn_s_wait_tensorcnt)
#define CDNA5_HAVE_TDM 1
#endif
#endif
#ifndef CDNA5_HAVE_TDM
#define CDNA5_HAVE_TDM 0
#endif

__device__ __forceinline__ unsigned short f32_to_bf16_rne(float f) {
    union { float f; unsigned u; } c; c.f = f;
    unsigned u = c.u;
    unsigned r = u + 0x7FFFu + ((u >> 16) & 1u);
    return (unsigned short)(r >> 16);
}

// ---------------------------------------------------------------------------
// TDM: DMA a 128-row x 32-col f32 tile (tile start = gsrc) of the
// (32768 x 1024) hidden_states tensor into LDS at lds_byte_off.
// D# built per CDNA5 ISA (2-D tile, groups 2/3 zero).
// ---------------------------------------------------------------------------
__device__ __forceinline__ void tdm_tile_load(const float* gsrc, unsigned lds_byte_off) {
#if CDNA5_HAVE_TDM
    unsigned long long ga = (unsigned long long)(const void*)gsrc;
    u32x4 g0 = { 1u,                                    // count=1, user desc
                 lds_byte_off,                          // lds_addr
                 (unsigned)ga,                          // global_addr[31:0]
                 (unsigned)((ga >> 32) & 0x01FFFFFFull) // global_addr[56:32]
                   | 0x80000000u };                     // type=2 ("image")
    i32x8 g1 = { 0x00020000,            // data_size=2 (4B); no pad/iterate/mcast
                 (int)(1024u << 16),    // tensor_dim0 = 1024 (low16 @ [63:48])
                 (int)0x80000000u,      // tensor_dim1 = 32768 (low16 @ [95:80])
                 (int)(32u << 16),      // tile_dim0 = 32 elements (one K chunk)
                 128,                   // tile_dim1 = 128 rows
                 1024,                  // tensor_dim0_stride = 1024
                 0, 0 };
    i32x4 z4 = { 0, 0, 0, 0 };
#if __clang_major__ >= 23
    i32x8 z8 = { 0, 0, 0, 0, 0, 0, 0, 0 };
    __builtin_amdgcn_tensor_load_to_lds(g0, g1, z4, z4, z8, 0);
#else
    __builtin_amdgcn_tensor_load_to_lds(g0, g1, z4, z4, 0);
#endif
#else
    (void)gsrc; (void)lds_byte_off;
#endif
}

// One K-chunk of the two 16x16x32 bf16 WMMAs (N tiles 0 and 1).
__device__ __forceinline__ void mma_step(const float* __restrict__ arow,
                                         const float* __restrict__ w0,
                                         const float* __restrict__ w1,
                                         int kc, int ka, int kb,
                                         v8f& acc0, v8f& acc1) {
    union { v16bf v; unsigned short s[16]; } A, B0, B1;
    #pragma unroll
    for (int e = 0; e < 8; ++e) {                    // A: 16x32 bf16, contiguous k = 16*half + e
        A.s[e]     = f32_to_bf16_rne(arow[ka + e]);
        A.s[8 + e] = f32_to_bf16_rne(arow[ka + 8 + e]);
    }
    #pragma unroll
    for (int e = 0; e < 16; ++e)                     // B: 32x16 bf16 layout
        B0.s[e] = f32_to_bf16_rne(w0[kc + kb + e]);
    if (w1) {
        #pragma unroll
        for (int e = 0; e < 16; ++e)
            B1.s[e] = f32_to_bf16_rne(w1[kc + kb + e]);
    } else {
        #pragma unroll
        for (int e = 0; e < 16; ++e) B1.s[e] = 0;
    }
    acc0 = __builtin_amdgcn_wmma_f32_16x16x32_bf16(false, A.v, false, B0.v,
                                                   (short)0, acc0, false, false);
    acc1 = __builtin_amdgcn_wmma_f32_16x16x32_bf16(false, A.v, false, B1.v,
                                                   (short)0, acc1, false, false);
}

// ---------------------------------------------------------------------------
// Kernel 1: emissions = hidden @ W^T + bias  via TDM->LDS double buffering +
// v_wmma_f32_16x16x32_bf16. Block = 256 threads (8 waves) = 128 output rows.
// ---------------------------------------------------------------------------
__global__ void __launch_bounds__(256)
crf_emissions_wmma(const float* __restrict__ hs,
                   const float* __restrict__ W,
                   const float* __restrict__ bias,
                   float* __restrict__ emis) {
    __shared__ float lds_a[2][128 * 32];              // 2 x 16 KB double buffer

    const int lane = threadIdx.x & 31;
    const int wave = threadIdx.x >> 5;
    const int r0b  = blockIdx.x * 128;                // block row base
    const int mrow = lane & 15;
    const int hiH  = lane >> 4;                       // lane half
    const int ka   = hiH ? 16 : 0;                    // A-frag K sub-base
    const int kb   = hiH ? 16 : 0;                    // B-frag K sub-base
    const int ncol = lane & 15;                       // N within tile
    const int n1   = 16 + ncol;

    const float* w0 = W + (size_t)ncol * Hn;          // tag = ncol (< 21 always)
    const float* w1 = (n1 < Tn) ? (W + (size_t)n1 * Hn) : nullptr;

    v8f acc0 = {}; v8f acc1 = {};

#if CDNA5_HAVE_TDM
    const unsigned ldsoff[2] = { (unsigned)(size_t)&lds_a[0][0],
                                 (unsigned)(size_t)&lds_a[1][0] };
    // Prologue: DMA chunk 0 into buffer 0 (wave 0 only; TENSORcnt is per-wave).
    if (threadIdx.x < 32)
        tdm_tile_load(hs + (size_t)r0b * Hn, ldsoff[0]);

    int pb = 0;
    for (int kc = 0; kc < Hn; kc += 32, pb ^= 1) {
        if (threadIdx.x < 32)
            __builtin_amdgcn_s_wait_tensorcnt((short)0);   // chunk kc landed
        __syncthreads();                                   // publish buf[pb]
        if (kc + 32 < Hn) {
            if (threadIdx.x < 32)                          // overlap next DMA
                tdm_tile_load(hs + (size_t)r0b * Hn + kc + 32, ldsoff[pb ^ 1]);
            __builtin_prefetch(w0 + kc + 32, 0, 0);        // global_prefetch_b8
        }
        const float* arow = &lds_a[pb][(wave * 16 + mrow) * 32];
        mma_step(arow, w0, w1, kc, ka, kb, acc0, acc1);
        __syncthreads();                                   // retire buf[pb]
    }
#else
    // Fallback: cooperative staged copy (no TDM builtin on this toolchain).
    for (int kc = 0; kc < Hn; kc += 32) {
        __syncthreads();
        for (int f = threadIdx.x; f < 128 * 32; f += 256) {
            const int row = f >> 5, col = f & 31;
            lds_a[0][f] = hs[(size_t)(r0b + row) * Hn + kc + col];
        }
        __syncthreads();
        const float* arow = &lds_a[0][(wave * 16 + mrow) * 32];
        mma_step(arow, w0, w1, kc, ka, kb, acc0, acc1);
    }
#endif

    // D layout: lane L, vgpr v -> M = v + 8*(L>=16), N = L&15
    const int   r0    = r0b + wave * 16;
    const int   mbase = hiH ? 8 : 0;
    const float b0    = bias[ncol];
    const float b1    = (n1 < Tn) ? bias[n1] : 0.f;
    #pragma unroll
    for (int v = 0; v < 8; ++v) {
        const size_t row = (size_t)(r0 + mbase + v);
        emis[row * TP + ncol] = acc0[v] + b0;
        emis[row * TP + n1]   = (n1 < Tn) ? (acc1[v] + b1) : -3.0e38f;  // pad
    }
}

// ---------------------------------------------------------------------------
// Kernel 2: CRF forward recursion + numerator. One wave per batch.
// ---------------------------------------------------------------------------
__global__ void crf_forward(const float* __restrict__ emis,
                            const int*   __restrict__ mask,
                            const int*   __restrict__ labels,
                            const float* __restrict__ trans,
                            const float* __restrict__ startt,
                            const float* __restrict__ endt,
                            float* __restrict__ partial) {
    const int b   = blockIdx.x;
    const int j   = threadIdx.x;          // 0..31, single wave
    const bool act = (j < Tn);

    float tcol[Tn];
    #pragma unroll
    for (int i = 0; i < Tn; ++i) tcol[i] = act ? trans[i * Tn + j] : -1.0e30f;

    const float* e = emis + (size_t)b * Sn * TP;
    float alpha = act ? (startt[j] + e[j]) : -1.0e30f;

    for (int t = 1; t < Sn; ++t) {
        float v[Tn];
        float m = -1.0e30f;
        #pragma unroll
        for (int i = 0; i < Tn; ++i) {
            const float ai = __shfl(alpha, i, 32);
            v[i] = ai + tcol[i];
            m = fmaxf(m, v[i]);
        }
        float s = 0.f;
        #pragma unroll
        for (int i = 0; i < Tn; ++i) s += __expf(v[i] - m);
        const float nxt = m + __logf(s) + e[t * TP + j];
        const int   mt  = mask[b * Sn + t];
        alpha = (mt && act) ? nxt : alpha;
    }

    float fin = act ? (alpha + endt[j]) : -1.0e30f;
    float m = fin;
    #pragma unroll
    for (int off = 16; off; off >>= 1) m = fmaxf(m, __shfl_xor(m, off, 32));
    float s = act ? __expf(fin - m) : 0.f;
    #pragma unroll
    for (int off = 16; off; off >>= 1) s += __shfl_xor(s, off, 32);
    const float logZ = m + __logf(s);

    if (j == 0) {
        int prev = labels[b * Sn];
        float num = startt[prev] + e[prev];
        for (int t = 1; t < Sn; ++t) {
            const int lt = labels[b * Sn + t];
            if (mask[b * Sn + t]) {
                num += trans[prev * Tn + lt] + e[t * TP + lt];
                prev = lt;
            }
        }
        num += endt[prev];
        partial[b] = logZ - num;
    }
}

__global__ void crf_loss_reduce(const float* __restrict__ partial, float* __restrict__ out) {
    if (threadIdx.x == 0) {
        float s = 0.f;
        for (int i = 0; i < Bn; ++i) s += partial[i];
        out[0] = s;
    }
}

// ---------------------------------------------------------------------------
// Kernel 3: Viterbi decode. One wave per batch; byte backpointers in LDS.
// ---------------------------------------------------------------------------
__global__ void crf_viterbi(const float* __restrict__ emis,
                            const int*   __restrict__ mask,
                            const float* __restrict__ trans,
                            const float* __restrict__ startt,
                            const float* __restrict__ endt,
                            float* __restrict__ out) {
    __shared__ unsigned char bp[Sn][32];   // 16 KB
    const int b   = blockIdx.x;
    const int j   = threadIdx.x;
    const bool act = (j < Tn);

    float tcol[Tn];
    #pragma unroll
    for (int i = 0; i < Tn; ++i) tcol[i] = act ? trans[i * Tn + j] : -1.0e30f;

    const float* e = emis + (size_t)b * Sn * TP;
    float score = act ? (startt[j] + e[j]) : -1.0e30f;

    for (int t = 1; t < Sn; ++t) {
        float best = -1.0e30f;
        int   bpi  = 0;
        #pragma unroll
        for (int i = 0; i < Tn; ++i) {
            const float ai = __shfl(score, i, 32);
            const float v  = ai + tcol[i];
            if (v > best) { best = v; bpi = i; }      // first-max like jnp.argmax
        }
        const float nxt = best + e[t * TP + j];
        const int   mt  = mask[b * Sn + t];
        if (mt && act) { score = nxt; bp[t][j] = (unsigned char)bpi; }
        else           {               bp[t][j] = (unsigned char)j; }
    }
    __syncthreads();

    float fin = act ? (score + endt[j]) : -1.0e30f;
    float m = fin; int mi = j;
    #pragma unroll
    for (int off = 16; off; off >>= 1) {
        const float om = __shfl_xor(m, off, 32);
        const int   oi = __shfl_xor(mi, off, 32);
        if (om > m || (om == m && oi < mi)) { m = om; mi = oi; }
    }
    int tag = __shfl(mi, 0, 32);

    if (j == 0) {
        for (int t = Sn - 1; t >= 1; --t) {
            out[1 + b * Sn + t] = mask[b * Sn + t] ? (float)tag : 0.f;
            tag = bp[t][tag];
        }
        out[1 + b * Sn] = mask[b * Sn] ? (float)tag : 0.f;
    }
}

// ---------------------------------------------------------------------------
extern "C" void kernel_launch(void* const* d_in, const int* in_sizes, int n_in,
                              void* d_out, int out_size, void* d_ws, size_t ws_size,
                              hipStream_t stream) {
    const float* hs     = (const float*)d_in[0];
    const int*   amask  = (const int*)  d_in[1];
    const int*   labels = (const int*)  d_in[2];
    const float* W      = (const float*)d_in[3];
    const float* bias   = (const float*)d_in[4];
    const float* trans  = (const float*)d_in[5];
    const float* startt = (const float*)d_in[6];
    const float* endt   = (const float*)d_in[7];
    float*       out    = (float*)d_out;

    float* emis    = (float*)d_ws;                    // 32768 * 32 floats = 4 MB
    float* partial = emis + (size_t)Bn * Sn * TP;     // 64 floats

    crf_emissions_wmma<<<256, 256, 0, stream>>>(hs, W, bias, emis);
    crf_forward       <<<Bn, 32, 0, stream>>>(emis, amask, labels, trans, startt, endt, partial);
    crf_loss_reduce   <<<1, 32, 0, stream>>>(partial, out);
    crf_viterbi       <<<Bn, 32, 0, stream>>>(emis, amask, trans, startt, endt, out);
}
